// MultiHeadAttention_59631325938240
// MI455X (gfx1250) — matrix operand order repaired and checked by reading, not compiled, not run on hardware
//
#include <hip/hip_runtime.h>
#include <hip/hip_bf16.h>

// ---------------------------------------------------------------------------
// MultiHeadAttention (faithful shape bug: attention is 16x16 over heads per
// token) for MI455X / gfx1250.
//
// Pipeline:
//   1. convert x (f32) -> xb (bf16)                         [16384 x 1024]
//   2. transpose+convert Wqkv -> WqkvT (bf16, [N][K])       [3072 x 1024]
//      transpose+convert Wout -> WoutT (bf16, [N][K])       [1024 x 1024]
//   3. GEMM1 (WMMA bf16, f32 acc): qkv = xb * Wqkv + bqkv   -> bf16 [16384 x 3072]
//   4. attention per token: 16x16 softmax over heads        -> ctx bf16 [16384 x 1024]
//   5. GEMM2 (WMMA bf16, f32 acc): out = ctx * Wout + bout  -> f32 d_out
//
// GEMM data movement (CDNA5-specific):
//   A tile : GLOBAL_LOAD_ASYNC_TO_LDS_B128 (per-lane, ASYNCcnt)
//   B tile : TENSOR_LOAD_TO_LDS (TDM, one issue per block by wave 0,
//            TENSORcnt, hardware LDS row padding to 80B stride)
//   Double-buffered LDS; compute overlaps next-tile transfers.
// ---------------------------------------------------------------------------

typedef __attribute__((ext_vector_type(16))) __bf16    v16bf;
typedef __attribute__((ext_vector_type(8)))  float     v8f;
typedef __attribute__((ext_vector_type(4)))  unsigned  v4u;
typedef __attribute__((ext_vector_type(8)))  int       v8i;
typedef __attribute__((ext_vector_type(4)))  int       v4i;

union Frag {
    v16bf v;
    float4 f[2];
};

// LDS byte address = low 32 bits of the generic pointer (gfx1250 LDS aperture
// occupies only the high 32 bits: isLDS <=> ADDR[63:32]=={SHARED_BASE,16'b0}).
__device__ __forceinline__ unsigned lds_addr32(const void* p)
{
    return (unsigned)(unsigned long long)(size_t)p;
}

// Copy 32 contiguous bytes global -> LDS without VGPR staging.
// INST_OFFSET applies to both the LDS and global address, so one base pair
// serves both halves. Tracked with ASYNCcnt.
__device__ __forceinline__ void async_copy_32B(unsigned ldsAddr, const void* gsrc)
{
    unsigned long long ga = (unsigned long long)(size_t)gsrc;
    asm volatile("global_load_async_to_lds_b128 %0, %1, off"
                 :: "v"(ldsAddr), "v"(ga) : "memory");
    asm volatile("global_load_async_to_lds_b128 %0, %1, off offset:16"
                 :: "v"(ldsAddr), "v"(ga) : "memory");
}

// TDM: load a 128-row x 32-bf16 tile (row stride = strideElems bf16 in global)
// into LDS with hardware padding: after every 16 DWORDs (one 64B row) skip
// 4 DWORDs -> LDS row stride 80B, matching sA/sB[128][40].
__device__ __forceinline__ void tdm_load_tile_128x32_bf16(
    unsigned ldsAddr, const void* gsrc, int strideElems)
{
#if __has_builtin(__builtin_amdgcn_tensor_load_to_lds)
    unsigned long long ga = (unsigned long long)(size_t)gsrc;
    // D# group 0: count=1 | lds_addr | global_addr[56:0] | type=2(image)
    v4u g0 = { 1u,
               ldsAddr,
               (unsigned)ga,
               (unsigned)(ga >> 32) | 0x80000000u };
    // D# group 1:
    //  w0: data_size=1(2B)<<16 | pad_enable<<20 | pad_interval=3(16DW)<<22
    //      | pad_amount=3(4DW)<<25
    //  tensor_dim0 = tensor_dim1 = 2^20 (large; exact-multiple tiles, no OOB)
    //  tile_dim0 = 32, tile_dim1 = 128, tensor_dim0_stride = strideElems
    v8i g1 = { (int)((1u << 16) | (1u << 20) | (3u << 22) | (3u << 25)),
               0,                           // abar=0 | tensor_dim0[15:0]=0
               16,                          // tensor_dim0[31:16]=16 | tensor_dim1[15:0]=0
               (int)(16u | (32u << 16)),    // tensor_dim1[31:16]=16 | tile_dim0=32
               128,                         // tile_dim1=128, tile_dim2=0
               strideElems,                 // tensor_dim0_stride[31:0]
               0, 0 };
    v4i z4 = { 0, 0, 0, 0 };
    v8i z8 = { 0, 0, 0, 0, 0, 0, 0, 0 };
    // 6-arg toolchain variant: (g0, g1, g2, g3, g4, cpol)
    __builtin_amdgcn_tensor_load_to_lds(g0, g1, z4, z4, z8, 0);
#endif
}

// ---------------------------------------------------------------------------
// Elementwise f32 -> bf16 convert
// ---------------------------------------------------------------------------
__global__ __launch_bounds__(256) void convert_f32_to_bf16(
    const float* __restrict__ in, __bf16* __restrict__ out, int n)
{
    int i = blockIdx.x * 256 + threadIdx.x;
    if (i < n) out[i] = (__bf16)in[i];
}

// ---------------------------------------------------------------------------
// Transpose + convert: in [K][N] f32 row-major -> out [N][K] bf16 row-major
// ---------------------------------------------------------------------------
__global__ __launch_bounds__(256) void transpose_convert_f32_to_bf16(
    const float* __restrict__ in, __bf16* __restrict__ out, int K, int N)
{
    int i = blockIdx.x * 256 + threadIdx.x;
    if (i < K * N) {
        int k = i / N;
        int n = i - k * N;
        out[(size_t)n * K + k] = (__bf16)in[i];
    }
}

// ---------------------------------------------------------------------------
// WMMA bf16 GEMM:  C[M][N] = A[M][K] * B[K][N] + bias[N]
//   A   : bf16 row-major [M][K]
//   BT  : bf16 row-major [N][K]   (B transposed)
//   out : bf16 (OUT_BF16) or f32
// Block tile 128x128, TILE_K = 32, 256 threads = 8 waves in a 2(M) x 4(N)
// grid; each wave owns a 64x32 sub-tile = 4x2 wmma accumulators.
// ---------------------------------------------------------------------------
#define TILE_K 32
#define LDS_STRIDE 40  // 32 + 8 bf16 pad (80B rows) to spread LDS banks

template <bool OUT_BF16>
__global__ __launch_bounds__(256) void gemm_bf16_wmma(
    const __bf16* __restrict__ A,
    const __bf16* __restrict__ BT,
    const float* __restrict__ bias,
    void* __restrict__ Cout,
    int M, int N, int K)
{
    __shared__ __align__(16) __bf16 sA[2][128][LDS_STRIDE];
    __shared__ __align__(16) __bf16 sB[2][128][LDS_STRIDE];

    const int tid   = threadIdx.x;
    const int lane  = tid & 31;
    const int wave  = tid >> 5;
    const int waveM = wave >> 2;   // 0..1
    const int waveN = wave & 3;    // 0..3
    const int mBlock = blockIdx.y * 128;
    const int nBlock = blockIdx.x * 128;

    // fragment lane decomposition
    const int fr = lane & 15;      // row (A) / col (B) within 16
    const int fh = lane >> 4;      // which K-half this lane holds

    v8f acc[4][2];
    v8f vzero = {0.f, 0.f, 0.f, 0.f, 0.f, 0.f, 0.f, 0.f};
#pragma unroll
    for (int mt = 0; mt < 4; ++mt)
#pragma unroll
        for (int nt = 0; nt < 2; ++nt)
            acc[mt][nt] = vzero;

    // A staging: each thread async-copies 32B (one half-row of the K-slab)
    const int ldRow  = tid >> 1;       // 0..127
    const int ldHalf = tid & 1;        // 0/1 -> K offset 16*ldHalf
    const __bf16* aBase = A + (size_t)(mBlock + ldRow) * K + ldHalf * 16;
    const unsigned ldsAslot[2] = {
        lds_addr32(&sA[0][ldRow][ldHalf * 16]),
        lds_addr32(&sA[1][ldRow][ldHalf * 16])
    };
    // B staging: whole 128x32 tile via one TDM descriptor (wave 0 only)
    const __bf16* bTileBase = BT + (size_t)nBlock * K;
    const unsigned ldsBslot[2] = {
        lds_addr32(&sB[0][0][0]),
        lds_addr32(&sB[1][0][0])
    };

    // prologue: kick off tile 0 into buffer 0
    async_copy_32B(ldsAslot[0], aBase);
    if (wave == 0)
        tdm_load_tile_128x32_bf16(ldsBslot[0], bTileBase, K);

    const int nSteps = K / TILE_K;
    for (int s = 0; s < nSteps; ++s) {
        const int cur = s & 1;
        const int nxt = cur ^ 1;
        const bool hasNext = (s + 1 < nSteps);

        __syncthreads();   // everyone done computing from buf[nxt] (prev iter)

        if (hasNext) {
            const int kb2 = (s + 1) * TILE_K;
            async_copy_32B(ldsAslot[nxt], aBase + kb2);
            if (wave == 0)
                tdm_load_tile_128x32_bf16(ldsBslot[nxt], bTileBase + kb2, K);
            asm volatile("s_wait_asynccnt 0x2" ::: "memory");  // buf[cur] A done
        } else {
            asm volatile("s_wait_asynccnt 0x0" ::: "memory");
        }
#if __has_builtin(__builtin_amdgcn_s_wait_tensorcnt)
        if (wave == 0) {
            if (hasNext) __builtin_amdgcn_s_wait_tensorcnt(1);  // buf[cur] B done
            else         __builtin_amdgcn_s_wait_tensorcnt(0);
        }
#endif
        __syncthreads();   // tile cur visible to all waves

        // --- load all fragments first, then issue 8 back-to-back WMMAs ---
        // B fragment (contiguous K order used here: lanes 0-15 hold K0..15,
        // lanes 16-31 hold K16..31, two K-values per VGPR)
        Frag bfr[2];
#pragma unroll
        for (int nt = 0; nt < 2; ++nt) {
            const __bf16* bp = &sB[cur][waveN * 32 + nt * 16 + fr][fh * 16];
            bfr[nt].f[0] = *(const float4*)bp;
            bfr[nt].f[1] = *(const float4*)(bp + 8);
        }
        // A fragment (same K order as B: lane half 0 holds
        // K0..15, lane half 1 holds K16..31)
        Frag af[4];
#pragma unroll
        for (int mt = 0; mt < 4; ++mt) {
            const __bf16* ap = &sA[cur][waveM * 64 + mt * 16 + fr][fh * 16];
            af[mt].f[0] = *(const float4*)ap;          // K = fh*16 .. fh*16+7
            af[mt].f[1] = *(const float4*)(ap + 8);    // K = fh*16+8 .. fh*16+15
        }
#pragma unroll
        for (int mt = 0; mt < 4; ++mt) {
#pragma unroll
            for (int nt = 0; nt < 2; ++nt) {
                acc[mt][nt] = __builtin_amdgcn_wmma_f32_16x16x32_bf16(
                    /*neg_a=*/false, af[mt].v,
                    /*neg_b=*/false, bfr[nt].v,
                    /*c_mod=*/(short)0, acc[mt][nt],
                    /*reuse_a=*/false, /*reuse_b=*/false);
            }
        }
    }

    // Epilogue: C/D f32 16x16 layout -> VGPR r, lane l:
    //   row = r + 8*(l>=16), col = l%16
#pragma unroll
    for (int nt = 0; nt < 2; ++nt) {
        const int col = nBlock + waveN * 32 + nt * 16 + fr;
        const float bv = bias[col];
#pragma unroll
        for (int mt = 0; mt < 4; ++mt) {
            const int row0 = mBlock + waveM * 64 + mt * 16 + fh * 8;
#pragma unroll
            for (int r = 0; r < 8; ++r) {
                float val = acc[mt][nt][r] + bv;
                size_t idx = (size_t)(row0 + r) * N + col;
                if (OUT_BF16) ((__bf16*)Cout)[idx] = (__bf16)val;
                else          ((float*)Cout)[idx]  = val;
            }
        }
    }
}

// ---------------------------------------------------------------------------
// Per-token 16x16 head-attention (the reference's shape bug).
// One wave per token; 4 tokens per 128-thread block.
//   logits[i][j] = scale * dot(q[i,:], k[j,:])   (i,j = head indices, 16x16)
//   attn = softmax_j(logits)
//   ctx[token, i*64+d] = sum_j attn[i][j] * v[j][d]
// ---------------------------------------------------------------------------
__global__ __launch_bounds__(128) void attn_heads16_kernel(
    const __bf16* __restrict__ qkv,   // [tokens][3072] = [q(1024) k(1024) v(1024)]
    __bf16* __restrict__ ctx)         // [tokens][1024]
{
    __shared__ float sQ[4][1024];
    __shared__ float sK[4][1024];
    __shared__ float sV[4][1024];
    __shared__ float sAttn[4][256];

    const int tid  = threadIdx.x;
    const int lane = tid & 31;
    const int w    = tid >> 5;                 // wave = token slot 0..3
    const int token = blockIdx.x * 4 + w;

    const __bf16* row = qkv + (size_t)token * 3072;
    for (int idx = lane; idx < 1024; idx += 32) {
        sQ[w][idx] = (float)row[idx];
        sK[w][idx] = (float)row[1024 + idx];
        sV[w][idx] = (float)row[2048 + idx];
    }
    __syncthreads();

    const int i = lane & 15;     // query-head row
    const int h = lane >> 4;     // half: this lane owns j = h*8 .. h*8+7
    const float scale = 0.125f;  // 1/sqrt(64)

    float lg[8];
#pragma unroll
    for (int jj = 0; jj < 8; ++jj) {
        const int j = h * 8 + jj;
        float s = 0.f;
#pragma unroll 8
        for (int d = 0; d < 64; ++d)
            s += sQ[w][i * 64 + d] * sK[w][j * 64 + d];
        lg[jj] = s * scale;
    }

    // softmax over j (16 values split across lane pairs l, l^16)
    float mx = lg[0];
#pragma unroll
    for (int jj = 1; jj < 8; ++jj) mx = fmaxf(mx, lg[jj]);
    mx = fmaxf(mx, __shfl_xor(mx, 16, 32));

    float ssum = 0.f;
#pragma unroll
    for (int jj = 0; jj < 8; ++jj) {
        lg[jj] = __expf(lg[jj] - mx);
        ssum += lg[jj];
    }
    ssum += __shfl_xor(ssum, 16, 32);
    const float inv = 1.f / ssum;

#pragma unroll
    for (int jj = 0; jj < 8; ++jj)
        sAttn[w][i * 16 + h * 8 + jj] = lg[jj] * inv;
    __syncthreads();

    // ctx row: lane (i, h) produces d = h*32 .. h*32+31 of output head i
    for (int d0 = 0; d0 < 32; ++d0) {
        const int d = h * 32 + d0;
        float a = 0.f;
#pragma unroll
        for (int j = 0; j < 16; ++j)
            a += sAttn[w][i * 16 + j] * sV[w][j * 64 + d];
        ctx[(size_t)token * 1024 + i * 64 + d] = (__bf16)a;
    }
}

// ---------------------------------------------------------------------------
// Host-side launcher
// ---------------------------------------------------------------------------
extern "C" void kernel_launch(void* const* d_in, const int* in_sizes, int n_in,
                              void* d_out, int out_size, void* d_ws, size_t ws_size,
                              hipStream_t stream)
{
    const float* x    = (const float*)d_in[0];   // [4,4096,1024]
    const float* Wqkv = (const float*)d_in[1];   // [1024,3072]
    const float* bqkv = (const float*)d_in[2];   // [3072]
    const float* Wout = (const float*)d_in[3];   // [1024,1024]
    const float* bout = (const float*)d_in[4];   // [1024]
    float* out = (float*)d_out;                  // [4,4096,1024] f32

    const int M  = 4 * 4096;   // 16384 tokens
    const int Kd = 1024;
    const int N1 = 3072;
    const int N2 = 1024;

    char* ws = (char*)d_ws;
    __bf16* xb    = (__bf16*)ws;  ws += (size_t)M  * Kd * sizeof(__bf16);
    __bf16* WqkvT = (__bf16*)ws;  ws += (size_t)N1 * Kd * sizeof(__bf16);
    __bf16* WoutT = (__bf16*)ws;  ws += (size_t)N2 * Kd * sizeof(__bf16);
    __bf16* qkvb  = (__bf16*)ws;  ws += (size_t)M  * N1 * sizeof(__bf16);
    __bf16* ctx   = (__bf16*)ws;  ws += (size_t)M  * Kd * sizeof(__bf16);

    // 1. convert inputs to bf16
    {
        int n = M * Kd;
        convert_f32_to_bf16<<<(n + 255) / 256, 256, 0, stream>>>(x, xb, n);
    }
    {
        int n = Kd * N1;
        transpose_convert_f32_to_bf16<<<(n + 255) / 256, 256, 0, stream>>>(Wqkv, WqkvT, Kd, N1);
    }
    {
        int n = Kd * N2;
        transpose_convert_f32_to_bf16<<<(n + 255) / 256, 256, 0, stream>>>(Wout, WoutT, Kd, N2);
    }

    // 2. qkv = xb * Wqkv + bqkv   (bf16 out)
    gemm_bf16_wmma<true><<<dim3(N1 / 128, M / 128), 256, 0, stream>>>(
        xb, WqkvT, bqkv, (void*)qkvb, M, N1, Kd);

    // 3. per-token 16x16 head attention
    attn_heads16_kernel<<<M / 4, 128, 0, stream>>>(qkvb, ctx);

    // 4. out = ctx * Wout + bout  (f32 out)
    gemm_bf16_wmma<false><<<dim3(N2 / 128, M / 128), 256, 0, stream>>>(
        ctx, WoutT, bout, (void*)out, M, N2, Kd);
}
